// GINE_GNN_56418690400934
// MI455X (gfx1250) — compile-verified, matrix operand order checked by reading, not run on hardware
//
#include <hip/hip_runtime.h>

#define N_NODES 50000
#define N_EDGES 800000
#define N_GRAPH 500
#define DIM     128
#define EDIM    7
#define LAYERS  5
#define NCLASS  10
#define BN_EPS  1e-5f

typedef __bf16 bf16_t;
typedef bf16_t v16bf __attribute__((ext_vector_type(16)));
typedef float  v8f   __attribute__((ext_vector_type(8)));
typedef unsigned int v4u __attribute__((ext_vector_type(4)));
typedef int    v8i   __attribute__((ext_vector_type(8)));
typedef int    v4i   __attribute__((ext_vector_type(4)));

// ---------------- workspace layout (in floats) ----------------
// x: N*D | agg: N*D | gsum: G*D | counts: 512 | h1: G*D | s1,t1,s2,t2: L*D each
// Wbf: 11 matrices of 128x128 bf16 pre-swizzled to WMMA B layout
//   slots: 2l = mlp_W1[l], 2l+1 = mlp_W2[l], 10 = head_W1   (~52.1 MB total)
#define ND_ELEMS   ((size_t)N_NODES * DIM)
#define X_OFF      ((size_t)0)
#define AGG_OFF    (ND_ELEMS)
#define GSUM_OFF   (2 * ND_ELEMS)
#define CNT_OFF    (GSUM_OFF + (size_t)N_GRAPH * DIM)
#define H1_OFF     (CNT_OFF + 512)
#define S1_OFF     (H1_OFF + (size_t)N_GRAPH * DIM)
#define T1_OFF     (S1_OFF + (size_t)LAYERS * DIM)
#define S2_OFF     (T1_OFF + (size_t)LAYERS * DIM)
#define T2_OFF     (S2_OFF + (size_t)LAYERS * DIM)
#define WBF_OFF    (T2_OFF + (size_t)LAYERS * DIM)   // byte offset divisible by 32

__device__ __forceinline__ void atomicAddF(float* p, float v) {
    __hip_atomic_fetch_add(p, v, __ATOMIC_RELAXED, __HIP_MEMORY_SCOPE_AGENT);
}

// ---------------- TDM: stage one 128x128 bf16 matrix (32 KB) into LDS ----------------
// D# per CDNA5 ISA: group0 = {count, lds_addr, global_addr, type=2},
// group1 = {data_size=2B, tensor_dim0=16384, tensor_dim1=1, tile=16384x1, stride=16384}.
// Tracked by TENSORcnt; pair with s_wait_tensorcnt. 6-arg builtin on this toolchain.
__device__ __forceinline__ void tdm_load_mat_to_lds(const bf16_t* gsrc, unsigned ldsByteOff) {
    unsigned long long ga = (unsigned long long)(const void*)gsrc;
    v4u g0;
    g0[0] = 1u;                                   // count=1 (valid user descriptor)
    g0[1] = ldsByteOff;                           // lds_addr (bytes)
    g0[2] = (unsigned)(ga & 0xFFFFFFFFu);         // global_addr[31:0]
    g0[3] = (unsigned)((ga >> 32) & 0x01FFFFFFu)  // global_addr[56:32]
            | (2u << 30);                         // type=2 ("image")
    v8i g1;
    g1[0] = (1 << 16);                            // workgroup_mask=0, data_size=1 (2 bytes)
    g1[1] = (int)(16384u << 16);                  // tensor_dim0[15:0]=16384 (bits 63:48)
    g1[2] = (1 << 16);                            // tensor_dim0 hi=0 | tensor_dim1=1
    g1[3] = (int)(16384u << 16);                  // tensor_dim1 hi=0 | tile_dim0=16384
    g1[4] = 1;                                    // tile_dim1=1, tile_dim2=0
    g1[5] = 16384;                                // tensor_dim0_stride lo32
    g1[6] = 0;                                    // stride hi / dim1_stride lo
    g1[7] = 0;
    v4i gz4 = {0, 0, 0, 0};                       // groups 2/3 unused (2-D tensor)
    v8i gz8 = {0, 0, 0, 0, 0, 0, 0, 0};           // extra group in 6-arg form
    __builtin_amdgcn_tensor_load_to_lds(g0, g1, gz4, gz4, gz8, 0);
}

// ---------------- trivial fills (vectorized) ----------------
__global__ void k_zero4(float4* __restrict__ p, int n4) {
    int i = blockIdx.x * 256 + threadIdx.x;
    if (i < n4) p[i] = make_float4(0.f, 0.f, 0.f, 0.f);
}

__global__ void k_init_x(const float4* __restrict__ emb4, float4* __restrict__ x4) {
    int i = blockIdx.x * 256 + threadIdx.x;
    if (i < (int)(ND_ELEMS / 4)) x4[i] = emb4[i & 31];   // 128/4 = 32 float4 per row
}

// ---------------- fold BatchNorm (+ linear bias) into scale/shift ----------------
__global__ void k_prep_bn(const float* __restrict__ lb1, const float* __restrict__ g1,
                          const float* __restrict__ b1,  const float* __restrict__ m1,
                          const float* __restrict__ v1,
                          const float* __restrict__ lb2, const float* __restrict__ g2,
                          const float* __restrict__ b2,  const float* __restrict__ m2,
                          const float* __restrict__ v2,
                          float* __restrict__ s1, float* __restrict__ t1,
                          float* __restrict__ s2, float* __restrict__ t2) {
    int i = blockIdx.x * 256 + threadIdx.x;
    if (i >= LAYERS * DIM) return;
    float sa = g1[i] * rsqrtf(v1[i] + BN_EPS);
    s1[i] = sa;
    t1[i] = (lb1[i] - m1[i]) * sa + b1[i];
    float sb = g2[i] * rsqrtf(v2[i] + BN_EPS);
    s2[i] = sb;
    t2[i] = (lb2[i] - m2[i]) * sb + b2[i];
}

// ---------------- pre-swizzle weights into WMMA B (32x16 bf16) layout ----------------
__global__ void k_swizzle(const float* __restrict__ W1, const float* __restrict__ W2,
                          const float* __restrict__ Wh, bf16_t* __restrict__ out) {
    int idx = blockIdx.x * 256 + threadIdx.x;
    if (idx >= 11 * DIM * DIM) return;
    int slot = idx >> 14;          // /16384
    int r    = idx & 16383;
    int e    = r & 15;             // element within v16bf (= v*2+h)
    int lane = (r >> 4) & 31;
    int nt   = (r >> 9) & 7;
    int kc   = r >> 12;
    int base = (e < 8) ? ((lane < 16) ? 0 : 8) : ((lane < 16) ? 16 : 24);
    int K  = kc * 32 + base + (e & 7);
    int Nc = nt * 16 + (lane & 15);
    const float* src;
    if (slot < 10) {
        int l = slot >> 1;
        src = (slot & 1) ? (W2 + (size_t)l * DIM * DIM) : (W1 + (size_t)l * DIM * DIM);
    } else {
        src = Wh;
    }
    out[idx] = (bf16_t)src[K * DIM + Nc];
}

// ---------------- edge message + scatter-add ----------------
__global__ void k_edge(const float* __restrict__ eattr, const int* __restrict__ srcI,
                       const int* __restrict__ dstI, const float* __restrict__ x,
                       const float* __restrict__ eW, const float* __restrict__ eB,
                       float* __restrict__ agg) {
    __shared__ __align__(16) float sW[EDIM * DIM];
    __shared__ __align__(16) float sB[DIM];
    int tid = threadIdx.x;
    for (int i = tid; i < EDIM * DIM; i += 256) sW[i] = eW[i];
    if (tid < DIM) sB[tid] = eB[tid];
    __syncthreads();

    int e = blockIdx.x * 8 + (tid >> 5);
    if (e >= N_EDGES) return;
    int lane = tid & 31;
    int c = lane * 4;

    float ea[EDIM];
#pragma unroll
    for (int k = 0; k < EDIM; ++k) ea[k] = eattr[(size_t)e * EDIM + k];

    float4 acc = *(const float4*)&sB[c];
#pragma unroll
    for (int k = 0; k < EDIM; ++k) {
        float v = ea[k];
        float4 w = *(const float4*)&sW[k * DIM + c];
        acc.x = fmaf(v, w.x, acc.x); acc.y = fmaf(v, w.y, acc.y);
        acc.z = fmaf(v, w.z, acc.z); acc.w = fmaf(v, w.w, acc.w);
    }
    int s = srcI[e], d = dstI[e];
    float4 xv = *(const float4*)(x + (size_t)s * DIM + c);   // coalesced 16B gather
    float m0 = fmaxf(xv.x + acc.x, 0.0f);
    float m1 = fmaxf(xv.y + acc.y, 0.0f);
    float m2 = fmaxf(xv.z + acc.z, 0.0f);
    float m3 = fmaxf(xv.w + acc.w, 0.0f);
    float* ap = agg + (size_t)d * DIM + c;
    atomicAddF(ap + 0, m0); atomicAddF(ap + 1, m1);
    atomicAddF(ap + 2, m2); atomicAddF(ap + 3, m3);
}

// ---------------- fused node update: (x+agg) @ W1 -> BN1/relu -> @ W2 -> BN2/relu ----------------
// 4 waves/block, 16 nodes/wave. Weights TDM-staged into LDS (W1, then W2 in the same
// buffer), overlapped with A-fragment construction. Transpose bounce buffer is bf16.
#define NODE_WAVES 4
#define TR_STRIDE  136   // halves; lanes land 4 banks apart -> conflict-free reads

struct NodeShared {
    bf16_t w[DIM * DIM];                       // 32 KB staged weight matrix
    bf16_t t[NODE_WAVES * 16 * TR_STRIDE];     // 17 KB bf16 transpose buffer
};

// pack one 8-float segment pair (b128 loads) into A-fragment halves
__device__ __forceinline__ void packA8(v16bf& a, int at, const float* p0, const float* p1) {
    float4 u0 = *(const float4*)p0;              // 32B-aligned segments
    float4 u1 = *(const float4*)(p0 + 4);
    float4 w0 = *(const float4*)p1;
    float4 w1 = *(const float4*)(p1 + 4);
    a[at + 0] = (bf16_t)(u0.x + w0.x); a[at + 1] = (bf16_t)(u0.y + w0.y);
    a[at + 2] = (bf16_t)(u0.z + w0.z); a[at + 3] = (bf16_t)(u0.w + w0.w);
    a[at + 4] = (bf16_t)(u1.x + w1.x); a[at + 5] = (bf16_t)(u1.y + w1.y);
    a[at + 6] = (bf16_t)(u1.z + w1.z); a[at + 7] = (bf16_t)(u1.w + w1.w);
}

__global__ void k_node(const float* __restrict__ x_in, const float* __restrict__ agg,
                       const bf16_t* __restrict__ W1s, const bf16_t* __restrict__ W2s,
                       const float* __restrict__ s1, const float* __restrict__ t1,
                       const float* __restrict__ s2, const float* __restrict__ t2,
                       float* __restrict__ x_out) {
    __shared__ NodeShared sh;
    const int tid  = threadIdx.x;
    const int wave = tid >> 5;
    const int lane = tid & 31;
    const int m    = lane & 15;
    const int hi   = lane >> 4;
    const int tileBase = blockIdx.x * (NODE_WAVES * 16) + wave * 16;
    const unsigned wOff = (unsigned)(size_t)((char*)sh.w - (char*)&sh); // LDS byte offset

    // kick off W1 DMA (one TDM issue per block; EXEC ignored by TDM)
    if (wave == 0) tdm_load_mat_to_lds(W1s, wOff);

    int rowL = tileBase + m;
    if (rowL >= N_NODES) rowL = N_NODES - 1;          // clamp; stores are guarded
    const float* xr = x_in + (size_t)rowL * DIM;
    const float* ar = agg  + (size_t)rowL * DIM;

    // A tile (16x128 bf16) in ISA A-layout, from global: h = x + agg  (overlaps DMA)
    v16bf a[4];
#pragma unroll
    for (int kc = 0; kc < 4; ++kc) {
        int b0 = kc * 32 + hi * 8;
        int b1 = b0 + 16;
        packA8(a[kc], 0, xr + b0, ar + b0);
        packA8(a[kc], 8, xr + b1, ar + b1);
    }

    __builtin_amdgcn_s_wait_tensorcnt((short)0);      // W1 resident
    __syncthreads();

    bf16_t* tl = sh.t + wave * (16 * TR_STRIDE);

    // GEMM1 (B from LDS) + folded BN1 + relu -> bf16 transpose buffer
#pragma unroll
    for (int nt = 0; nt < 8; ++nt) {
        v8f acc = {0.f, 0.f, 0.f, 0.f, 0.f, 0.f, 0.f, 0.f};
#pragma unroll
        for (int kc = 0; kc < 4; ++kc) {
            v16bf b = *(const v16bf*)(sh.w + (((kc * 8 + nt) * 32 + lane) << 4));
            acc = __builtin_amdgcn_wmma_f32_16x16x32_bf16(false, a[kc], false, b,
                                                          (short)0, acc, false, false);
        }
        int nc = nt * 16 + m;
        float sc = s1[nc], tc = t1[nc];
#pragma unroll
        for (int r = 0; r < 8; ++r) {
            float y = fmaxf(fmaf(acc[r], sc, tc), 0.0f);
            tl[(r + hi * 8) * TR_STRIDE + nc] = (bf16_t)y;
        }
    }
    __syncthreads();                                   // all B reads of W1 done

    // overwrite weight buffer with W2 while we reload A from the transpose buffer
    if (wave == 0) tdm_load_mat_to_lds(W2s, wOff);

#pragma unroll
    for (int kc = 0; kc < 4; ++kc) {
        int b0 = kc * 32 + hi * 8;
        int b1 = b0 + 16;
        const bf16_t* hr = tl + m * TR_STRIDE;
#pragma unroll
        for (int j = 0; j < 8; ++j) a[kc][j]     = hr[b0 + j];
#pragma unroll
        for (int j = 0; j < 8; ++j) a[kc][8 + j] = hr[b1 + j];
    }

    __builtin_amdgcn_s_wait_tensorcnt((short)0);      // W2 resident
    __syncthreads();

    // GEMM2 (B from LDS) + folded BN2 + relu -> x_out (in place, tiles disjoint)
#pragma unroll
    for (int nt = 0; nt < 8; ++nt) {
        v8f acc = {0.f, 0.f, 0.f, 0.f, 0.f, 0.f, 0.f, 0.f};
#pragma unroll
        for (int kc = 0; kc < 4; ++kc) {
            v16bf b = *(const v16bf*)(sh.w + (((kc * 8 + nt) * 32 + lane) << 4));
            acc = __builtin_amdgcn_wmma_f32_16x16x32_bf16(false, a[kc], false, b,
                                                          (short)0, acc, false, false);
        }
        int nc = nt * 16 + m;
        float sc = s2[nc], tc = t2[nc];
#pragma unroll
        for (int r = 0; r < 8; ++r) {
            float y = fmaxf(fmaf(acc[r], sc, tc), 0.0f);
            int node = tileBase + r + hi * 8;
            if (node < N_NODES) x_out[(size_t)node * DIM + nc] = y;
        }
    }
}

// ---------------- graph mean-pool (atomic accumulate) ----------------
__global__ void k_pool(const float* __restrict__ x, const int* __restrict__ batch,
                       float* __restrict__ gsum, float* __restrict__ counts) {
    int idx = blockIdx.x * 256 + threadIdx.x;
    int n = idx >> 5;
    if (n >= N_NODES) return;
    int lane = idx & 31;
    int c = lane * 4;
    int b = batch[n];
    float4 xv = *(const float4*)(x + (size_t)n * DIM + c);
    float* gp = gsum + (size_t)b * DIM + c;
    atomicAddF(gp + 0, xv.x); atomicAddF(gp + 1, xv.y);
    atomicAddF(gp + 2, xv.z); atomicAddF(gp + 3, xv.w);
    if (lane == 0) atomicAddF(&counts[b], 1.0f);
}

// ---------------- head layer 1: relu((gsum/cnt) @ head_W1 + b1) via WMMA ----------------
__global__ void k_head(const float* __restrict__ gsum, const float* __restrict__ counts,
                       const bf16_t* __restrict__ Whs, const float* __restrict__ hb1,
                       float* __restrict__ h1) {
    const int tid  = threadIdx.x;
    const int wave = tid >> 5;
    const int lane = tid & 31;
    const int m    = lane & 15;
    const int hi   = lane >> 4;
    const int tileBase = blockIdx.x * (NODE_WAVES * 16) + wave * 16;

    int rowL = tileBase + m;
    if (rowL >= N_GRAPH) rowL = N_GRAPH - 1;
    float inv = 1.0f / fmaxf(counts[rowL], 1.0f);
    const float* gr = gsum + (size_t)rowL * DIM;

    v16bf a[4];
#pragma unroll
    for (int kc = 0; kc < 4; ++kc) {
        int b0 = kc * 32 + hi * 8;
        int b1 = b0 + 16;
        float4 u0 = *(const float4*)(gr + b0);
        float4 u1 = *(const float4*)(gr + b0 + 4);
        float4 w0 = *(const float4*)(gr + b1);
        float4 w1 = *(const float4*)(gr + b1 + 4);
        a[kc][0]  = (bf16_t)(u0.x * inv); a[kc][1]  = (bf16_t)(u0.y * inv);
        a[kc][2]  = (bf16_t)(u0.z * inv); a[kc][3]  = (bf16_t)(u0.w * inv);
        a[kc][4]  = (bf16_t)(u1.x * inv); a[kc][5]  = (bf16_t)(u1.y * inv);
        a[kc][6]  = (bf16_t)(u1.z * inv); a[kc][7]  = (bf16_t)(u1.w * inv);
        a[kc][8]  = (bf16_t)(w0.x * inv); a[kc][9]  = (bf16_t)(w0.y * inv);
        a[kc][10] = (bf16_t)(w0.z * inv); a[kc][11] = (bf16_t)(w0.w * inv);
        a[kc][12] = (bf16_t)(w1.x * inv); a[kc][13] = (bf16_t)(w1.y * inv);
        a[kc][14] = (bf16_t)(w1.z * inv); a[kc][15] = (bf16_t)(w1.w * inv);
    }

#pragma unroll
    for (int nt = 0; nt < 8; ++nt) {
        v8f acc = {0.f, 0.f, 0.f, 0.f, 0.f, 0.f, 0.f, 0.f};
#pragma unroll
        for (int kc = 0; kc < 4; ++kc) {
            v16bf b = *(const v16bf*)(Whs + (((kc * 8 + nt) * 32 + lane) << 4));
            acc = __builtin_amdgcn_wmma_f32_16x16x32_bf16(false, a[kc], false, b,
                                                          (short)0, acc, false, false);
        }
        int nc = nt * 16 + m;
        float bias = hb1[nc];
#pragma unroll
        for (int r = 0; r < 8; ++r) {
            float y = fmaxf(acc[r] + bias, 0.0f);
            int row = tileBase + r + hi * 8;
            if (row < N_GRAPH) h1[(size_t)row * DIM + nc] = y;
        }
    }
}

// ---------------- head layer 2: h1 @ head_W2 + b2 (tiny, scalar) ----------------
__global__ void k_out(const float* __restrict__ h1, const float* __restrict__ W2,
                      const float* __restrict__ b2, float* __restrict__ out) {
    int idx = blockIdx.x * 256 + threadIdx.x;
    if (idx >= N_GRAPH * NCLASS) return;
    int r = idx / NCLASS, c = idx % NCLASS;
    float acc = b2[c];
    const float* hr = h1 + (size_t)r * DIM;
#pragma unroll 16
    for (int k = 0; k < DIM; ++k) acc = fmaf(hr[k], W2[k * NCLASS + c], acc);
    out[idx] = acc;
}

extern "C" void kernel_launch(void* const* d_in, const int* in_sizes, int n_in,
                              void* d_out, int out_size, void* d_ws, size_t ws_size,
                              hipStream_t stream) {
    (void)in_sizes; (void)n_in; (void)out_size; (void)ws_size;
    const float* edge_attr = (const float*)d_in[0];
    const int*   edge_idx  = (const int*)d_in[1];
    const int*   batch     = (const int*)d_in[2];
    const float* node_emb  = (const float*)d_in[3];
    const float* edge_W    = (const float*)d_in[4];
    const float* edge_b    = (const float*)d_in[5];
    const float* mlp_W1    = (const float*)d_in[6];
    const float* mlp_b1    = (const float*)d_in[7];
    const float* bn1_g     = (const float*)d_in[8];
    const float* bn1_b     = (const float*)d_in[9];
    const float* bn1_m     = (const float*)d_in[10];
    const float* bn1_v     = (const float*)d_in[11];
    const float* mlp_W2    = (const float*)d_in[12];
    const float* mlp_b2    = (const float*)d_in[13];
    const float* bn2_g     = (const float*)d_in[14];
    const float* bn2_b     = (const float*)d_in[15];
    const float* bn2_m     = (const float*)d_in[16];
    const float* bn2_v     = (const float*)d_in[17];
    const float* head_W1   = (const float*)d_in[18];
    const float* head_b1   = (const float*)d_in[19];
    const float* head_W2   = (const float*)d_in[20];
    const float* head_b2   = (const float*)d_in[21];

    float* ws     = (float*)d_ws;
    float* x      = ws + X_OFF;
    float* agg    = ws + AGG_OFF;
    float* gsum   = ws + GSUM_OFF;
    float* counts = ws + CNT_OFF;
    float* h1     = ws + H1_OFF;
    float* s1     = ws + S1_OFF;
    float* t1     = ws + T1_OFF;
    float* s2     = ws + S2_OFF;
    float* t2     = ws + T2_OFF;
    bf16_t* Wbf   = (bf16_t*)(ws + WBF_OFF);

    const int* srcI = edge_idx;
    const int* dstI = edge_idx + N_EDGES;

    k_prep_bn<<<(LAYERS * DIM + 255) / 256, 256, 0, stream>>>(
        mlp_b1, bn1_g, bn1_b, bn1_m, bn1_v,
        mlp_b2, bn2_g, bn2_b, bn2_m, bn2_v, s1, t1, s2, t2);
    k_swizzle<<<(11 * DIM * DIM + 255) / 256, 256, 0, stream>>>(mlp_W1, mlp_W2, head_W1, Wbf);
    k_init_x<<<((int)(ND_ELEMS / 4) + 255) / 256, 256, 0, stream>>>(
        (const float4*)node_emb, (float4*)x);

    for (int l = 0; l < LAYERS; ++l) {
        k_zero4<<<((int)(ND_ELEMS / 4) + 255) / 256, 256, 0, stream>>>(
            (float4*)agg, (int)(ND_ELEMS / 4));
        k_edge<<<(N_EDGES + 7) / 8, 256, 0, stream>>>(
            edge_attr, srcI, dstI, x,
            edge_W + (size_t)l * EDIM * DIM, edge_b + (size_t)l * DIM, agg);
        k_node<<<(N_NODES + NODE_WAVES * 16 - 1) / (NODE_WAVES * 16), NODE_WAVES * 32, 0, stream>>>(
            x, agg,
            Wbf + (size_t)(2 * l) * DIM * DIM, Wbf + (size_t)(2 * l + 1) * DIM * DIM,
            s1 + l * DIM, t1 + l * DIM, s2 + l * DIM, t2 + l * DIM, x);
    }

    k_zero4<<<(((N_GRAPH * DIM + 512) / 4) + 255) / 256, 256, 0, stream>>>(
        (float4*)gsum, (N_GRAPH * DIM + 512) / 4);
    k_pool<<<(N_NODES * 32 + 255) / 256, 256, 0, stream>>>(x, batch, gsum, counts);
    k_head<<<(N_GRAPH + NODE_WAVES * 16 - 1) / (NODE_WAVES * 16), NODE_WAVES * 32, 0, stream>>>(
        gsum, counts, Wbf + (size_t)10 * DIM * DIM, head_b1, h1);
    k_out<<<(N_GRAPH * NCLASS + 255) / 256, 256, 0, stream>>>(h1, head_W2, head_b2, (float*)d_out);
}
